// GraphSAGEPlusPlusDA_8830452760803
// MI455X (gfx1250) — matrix operand order repaired and checked by reading, not compiled, not run on hardware
//
#include <hip/hip_runtime.h>
#include <hip/hip_bf16.h>
#include <math.h>

#define N0 200000
#define N1 50000
#define N2 10000
#define D  256
#define C  40
#define DEG 10

typedef __attribute__((ext_vector_type(16))) __bf16 v16bf;
typedef __attribute__((ext_vector_type(8)))  __bf16 v8bf;
typedef __attribute__((ext_vector_type(8)))  float  v8f;

__device__ inline __bf16 f2bf(float f) {
    union { float f; unsigned u; } v; v.f = f;
    unsigned r = v.u + 0x7FFFu + ((v.u >> 16) & 1u);   // RNE
    unsigned short h = (unsigned short)(r >> 16);
    return __builtin_bit_cast(__bf16, h);
}
__device__ inline float bf2f(__bf16 b) {
    unsigned short h = __builtin_bit_cast(unsigned short, b);
    union { unsigned u; float f; } v; v.u = ((unsigned)h) << 16;
    return v.f;
}

// ---------------- f32 -> bf16 convert, 8 elements/thread ----------------
__global__ void cvt_bf16_kernel(const float* __restrict__ in, __bf16* __restrict__ out, int n8) {
    int i = blockIdx.x * blockDim.x + threadIdx.x;
    if (i < n8) {
        v8f v = *(const v8f*)(in + (size_t)i * 8);
        v8bf o;
#pragma unroll
        for (int j = 0; j < 8; ++j) o[j] = f2bf(v[j]);
        *(v8bf*)(out + (size_t)i * 8) = o;
    }
}

// ---------------- W (KxN f32) -> Wt (NxK bf16) ----------------
__global__ void wt_kernel(const float* __restrict__ W, __bf16* __restrict__ Wt) {
    __shared__ float tile[16][17];
    int kb = blockIdx.x * 16, nb = blockIdx.y * 16;
    tile[threadIdx.y][threadIdx.x] = W[(size_t)(kb + threadIdx.y) * D + nb + threadIdx.x];
    __syncthreads();
    Wt[(size_t)(nb + threadIdx.y) * D + kb + threadIdx.x] = f2bf(tile[threadIdx.x][threadIdx.y]);
}

// ---------------- layer-0 aggregation: mean + max in one gather pass ----------------
// One wave per target. Edge indices are wave-uniform -> scalar loads, no LDS/barrier.
// Each lane owns 8 channels: 16-byte bf16 loads from the L2-resident bf16 copy of x.
__global__ __launch_bounds__(32) void agg0_kernel(
    const __bf16* __restrict__ xb, const int* __restrict__ row,
    __bf16* __restrict__ am, __bf16* __restrict__ ax) {
    int t = blockIdx.x;
    int c = threadIdx.x * 8;
    float s[8] = {};
    float m[8];
#pragma unroll
    for (int j = 0; j < 8; ++j) m[j] = -INFINITY;
#pragma unroll
    for (int e = 0; e < DEG; ++e) {
        int id = row[t * DEG + e];                       // uniform -> s_load
        if (e + 1 < DEG) __builtin_prefetch(xb + (size_t)row[t * DEG + e + 1] * D + c, 0, 0);
        v8bf v = *(const v8bf*)(xb + (size_t)id * D + c);
#pragma unroll
        for (int j = 0; j < 8; ++j) {
            float f = bf2f(v[j]);
            s[j] += f; m[j] = fmaxf(m[j], f);
        }
    }
    v8bf om_, ox_;
#pragma unroll
    for (int j = 0; j < 8; ++j) { om_[j] = f2bf(s[j] * (1.0f / DEG)); ox_[j] = f2bf(m[j]); }
    *(v8bf*)(am + (size_t)t * D + c) = om_;
    *(v8bf*)(ax + (size_t)t * D + c) = ox_;
}

// ---------------- layer-1 aggregation (two bf16 source matrices, one pass) ----------------
__global__ __launch_bounds__(32) void agg1_kernel(
    const __bf16* __restrict__ hm, const __bf16* __restrict__ hx,
    const int* __restrict__ row,
    __bf16* __restrict__ am, __bf16* __restrict__ ax) {
    int t = blockIdx.x;
    int c = threadIdx.x * 8;
    float s[8] = {};
    float m[8];
#pragma unroll
    for (int j = 0; j < 8; ++j) m[j] = -INFINITY;
#pragma unroll
    for (int e = 0; e < DEG; ++e) {
        int id = row[t * DEG + e];                       // uniform -> s_load
        size_t o = (size_t)id * D + c;
        v8bf vm = *(const v8bf*)(hm + o);
        v8bf vx = *(const v8bf*)(hx + o);
#pragma unroll
        for (int j = 0; j < 8; ++j) {
            s[j] += bf2f(vm[j]);
            m[j] = fmaxf(m[j], bf2f(vx[j]));
        }
    }
    v8bf om_, ox_;
#pragma unroll
    for (int j = 0; j < 8; ++j) { om_[j] = f2bf(s[j] * (1.0f / DEG)); ox_[j] = f2bf(m[j]); }
    *(v8bf*)(am + (size_t)t * D + c) = om_;
    *(v8bf*)(ax + (size_t)t * D + c) = ox_;
}

// ---------------- one K=256 accumulation pass of the dual GEMM ----------------
// Straight-line helper so A/Bt stay traceable kernarg pointers -> global_load_* selection.
// All 8 B fragments land in distinct locals before the 8-WMMA burst so the scheduler
// overlaps the next k-step's loads with the current WMMAs (partial s_wait_loadcnt).
__device__ __forceinline__ void gemm_pass_k256(
    const __bf16* __restrict__ A, const __bf16* __restrict__ Bt,
    size_t arow, int colBase, int l15, int lhi, v8f acc[8]) {
#pragma unroll
    for (int k = 0; k < D; k += 32) {
        // A fragment: ISA 16-bit A layout — lane m=l15 holds K [kb,kb+8) and [kb+16,kb+24), kb = k + lhi*8
        v16bf af;
        *(v8bf*)&af       = *(const v8bf*)(A + arow + k + lhi * 8);
        *((v8bf*)&af + 1) = *(const v8bf*)(A + arow + k + lhi * 8 + 16);
        v16bf b[8];
#pragma unroll
        for (int t = 0; t < 8; ++t) {
            int n = colBase + t * 16 + l15;
            *(v8bf*)&b[t]       = *(const v8bf*)(Bt + (size_t)n * D + k + lhi * 8);
            *((v8bf*)&b[t] + 1) = *(const v8bf*)(Bt + (size_t)n * D + k + lhi * 8 + 16);
        }
#pragma unroll
        for (int t = 0; t < 8; ++t)
            acc[t] = __builtin_amdgcn_wmma_f32_16x16x32_bf16(
                false, af, false, b[t], (short)0, acc[t], false, false);
    }
}

// ---------------- fused dual GEMM: out = act(A1@B1 + A2@B2 + bias) ----------------
// A1,A2: M x 256 bf16 row-major.  B1t,B2t: 256 x 256 bf16, TRANSPOSED (N x K).
// Workgroup = 4 waves, tile 64 rows; grid.y in {0,1} selects the 128-column half.
// Each wave: 16 rows x 128 cols = 8 WMMA accumulators.
template <bool RELU, bool OUT_BF16>
__global__ __launch_bounds__(128) void dual_gemm_kernel(
    const __bf16* __restrict__ A1, const __bf16* __restrict__ B1t,
    const __bf16* __restrict__ A2, const __bf16* __restrict__ B2t,
    const float* __restrict__ bias, int M,
    __bf16* __restrict__ outb, float* __restrict__ outf)
{
    const int lane = threadIdx.x & 31;
    const int wave = threadIdx.x >> 5;
    const int l15  = lane & 15;
    const int lhi  = lane >> 4;
    const int rowBase = blockIdx.x * 64 + wave * 16;
    const int colBase = blockIdx.y * 128;

    int mrow = rowBase + l15;
    if (mrow > M - 1) mrow = M - 1;           // clamp: loads stay in-bounds, EXEC stays full
    const size_t arow = (size_t)mrow * D;

    v8f acc[8] = {};

    gemm_pass_k256(A1, B1t, arow, colBase, l15, lhi, acc);
    gemm_pass_k256(A2, B2t, arow, colBase, l15, lhi, acc);

#pragma unroll
    for (int t = 0; t < 8; ++t) {
#pragma unroll
        for (int r = 0; r < 8; ++r) {
            int row = rowBase + r + lhi * 8;   // C/D layout: lanes 16-31 hold M = r+8
            int col = colBase + t * 16 + l15;
            if (row < M) {
                float v = acc[t][r] + bias[col];
                if (RELU) v = fmaxf(v, 0.f);
                if (OUT_BF16) outb[(size_t)row * D + col] = f2bf(v);
                else          outf[(size_t)row * D + col] = v;
            }
        }
    }
}

// ---------------- post: concat(o_m,o_x) @ W_post + b_post, then log_softmax ----------------
__global__ __launch_bounds__(64) void post_kernel(
    const float* __restrict__ om, const float* __restrict__ ox,
    const float* __restrict__ Wp, const float* __restrict__ bp,
    float* __restrict__ out) {
    __shared__ float row[2 * D];
    __shared__ float logit[C];
    __shared__ float red[2];
    int i = blockIdx.x;
    for (int k = threadIdx.x; k < D; k += blockDim.x) {
        row[k]     = om[(size_t)i * D + k];
        row[D + k] = ox[(size_t)i * D + k];
    }
    __syncthreads();
    if (threadIdx.x < C) {
        float a = bp[threadIdx.x];
#pragma unroll 8
        for (int k = 0; k < 2 * D; ++k) a += row[k] * Wp[(size_t)k * C + threadIdx.x];
        logit[threadIdx.x] = a;
    }
    __syncthreads();
    if (threadIdx.x == 0) {
        float mx = -INFINITY;
        for (int c = 0; c < C; ++c) mx = fmaxf(mx, logit[c]);
        float s = 0.f;
        for (int c = 0; c < C; ++c) s += __expf(logit[c] - mx);
        red[0] = mx; red[1] = __logf(s);
    }
    __syncthreads();
    if (threadIdx.x < C) out[(size_t)i * C + threadIdx.x] = logit[threadIdx.x] - red[0] - red[1];
}

extern "C" void kernel_launch(void* const* d_in, const int* in_sizes, int n_in,
                              void* d_out, int out_size, void* d_ws, size_t ws_size,
                              hipStream_t stream) {
    const float* x     = (const float*)d_in[0];
    const int*   ei0   = (const int*)d_in[1];     // row0 = ei0[0 : N1*DEG]
    const int*   ei1   = (const int*)d_in[2];     // row1 = ei1[0 : N2*DEG]
    const float* Wl_m0 = (const float*)d_in[3];
    const float* Wr_m0 = (const float*)d_in[4];
    const float* b_m0  = (const float*)d_in[5];
    const float* Wl_x0 = (const float*)d_in[6];
    const float* Wr_x0 = (const float*)d_in[7];
    const float* b_x0  = (const float*)d_in[8];
    const float* Wl_m1 = (const float*)d_in[9];
    const float* Wr_m1 = (const float*)d_in[10];
    const float* b_m1  = (const float*)d_in[11];
    const float* Wl_x1 = (const float*)d_in[12];
    const float* Wr_x1 = (const float*)d_in[13];
    const float* b_x1  = (const float*)d_in[14];
    const float* Wpost = (const float*)d_in[15];
    const float* bpost = (const float*)d_in[16];
    float* out = (float*)d_out;

    // ---- carve workspace (all 256B-aligned) ----
    char* p = (char*)d_ws;
    auto carve = [&](size_t bytes) { void* r = (void*)p; p += (bytes + 255) & ~(size_t)255; return r; };
    __bf16* xbf    = (__bf16*)carve((size_t)N0 * D * 2);   // bf16(x): gather source + GEMM A2 prefix
    __bf16* aggm0  = (__bf16*)carve((size_t)N1 * D * 2);
    __bf16* aggx0  = (__bf16*)carve((size_t)N1 * D * 2);
    __bf16* hm     = (__bf16*)carve((size_t)N1 * D * 2);
    __bf16* hx     = (__bf16*)carve((size_t)N1 * D * 2);
    __bf16* aggm1  = (__bf16*)carve((size_t)N2 * D * 2);
    __bf16* aggx1  = (__bf16*)carve((size_t)N2 * D * 2);
    float*  om     = (float*)carve((size_t)N2 * D * 4);
    float*  oxv    = (float*)carve((size_t)N2 * D * 4);
    __bf16* Wt[8];
    const float* Wsrc[8] = { Wl_m0, Wr_m0, Wl_x0, Wr_x0, Wl_m1, Wr_m1, Wl_x1, Wr_x1 };
    for (int i = 0; i < 8; ++i) Wt[i] = (__bf16*)carve((size_t)D * D * 2);

    // ---- prep: bf16 copy of ALL of x (halves gather traffic, makes it L2-resident) ----
    {
        int n8 = (N0 * D) / 8;
        cvt_bf16_kernel<<<(n8 + 255) / 256, 256, 0, stream>>>(x, xbf, n8);
    }
    for (int i = 0; i < 8; ++i)
        wt_kernel<<<dim3(D / 16, D / 16), dim3(16, 16), 0, stream>>>(Wsrc[i], Wt[i]);

    // ---- layer 0 ----
    agg0_kernel<<<N1, 32, 0, stream>>>(xbf, ei0, aggm0, aggx0);
    {
        dim3 g((N1 + 63) / 64, 2);
        dual_gemm_kernel<true, true><<<g, 128, 0, stream>>>(aggm0, Wt[0], xbf, Wt[1], b_m0, N1, hm, nullptr);
        dual_gemm_kernel<true, true><<<g, 128, 0, stream>>>(aggx0, Wt[2], xbf, Wt[3], b_x0, N1, hx, nullptr);
    }

    // ---- layer 1 ----
    agg1_kernel<<<N2, 32, 0, stream>>>(hm, hx, ei1, aggm1, aggx1);
    {
        dim3 g((N2 + 63) / 64, 2);
        dual_gemm_kernel<false, false><<<g, 128, 0, stream>>>(aggm1, Wt[4], xbf, Wt[5], b_m1, N2, nullptr, om);
        dual_gemm_kernel<false, false><<<g, 128, 0, stream>>>(aggx1, Wt[6], xbf, Wt[7], b_x1, N2, nullptr, oxv);
    }

    // ---- post + log_softmax ----
    post_kernel<<<N2, 64, 0, stream>>>(om, oxv, Wpost, bpost, out);
}
